// VCN_61108794688189
// MI455X (gfx1250) — matrix operand order repaired and checked by reading, not compiled, not run on hardware
//
#include <hip/hip_runtime.h>
#include <hip/hip_bf16.h>
#include <cmath>

typedef __attribute__((ext_vector_type(16))) _Float16 v16h;
typedef __attribute__((ext_vector_type(8)))  _Float16 v8h;
typedef __attribute__((ext_vector_type(8)))  float    v8f;

#define MD_  4
#define U_   9
#define NUV  81
#define C_   128
#define F_   16
#define H_   64
#define W_   64
#define B_   2
#define HW_  (H_ * W_)
#define PAD_ 72            // H_ + 2*MD_ = zero-padded tar tile edge

// ---------------------------------------------------------------------------
// Pre-kernel A: ref (B,C,H,W) f32 -> (B,H,W,C) f16 channel-last in ws.
// ---------------------------------------------------------------------------
extern "C" __global__ void vcn_cvt_ref_kernel(const float* __restrict__ ref,
                                              _Float16* __restrict__ ref16) {
  const int idx = blockIdx.x * blockDim.x + threadIdx.x;
  if (idx >= B_ * HW_ * C_) return;
  const int c = idx & 127;
  const int p = (idx >> 7) & (HW_ - 1);
  const int b = idx >> 19;
  ref16[idx] = (_Float16)ref[((size_t)(b * C_ + c)) * HW_ + p];
}

// ---------------------------------------------------------------------------
// Pre-kernel B: tar (B,C,H,W) f32 -> (B,72,72,C) f16 channel-last with a
// zero MD_-halo, so the shift loop needs no bounds handling at all.
// ---------------------------------------------------------------------------
extern "C" __global__ void vcn_cvt_tar_kernel(const float* __restrict__ tar,
                                              _Float16* __restrict__ tar16) {
  const int idx = blockIdx.x * blockDim.x + threadIdx.x;
  if (idx >= B_ * PAD_ * PAD_ * C_) return;
  const int c = idx & 127;
  int t = idx >> 7;
  const int xx = t % PAD_; t /= PAD_;
  const int yy = t % PAD_;
  const int b  = t / PAD_;
  const int ty = yy - MD_, tx = xx - MD_;
  const bool ok = (ty >= 0) && (ty < H_) && (tx >= 0) && (tx < W_);
  const float v = ok ? tar[((size_t)(b * C_ + c) * H_ + ty) * W_ + tx] : 0.0f;
  tar16[idx] = (_Float16)v;
}

// ---------------------------------------------------------------------------
// Kernel 1: fused corrf + 1x1 projection (WMMA) + flow_reg.
// One wave per (b, y0, 16-wide x tile). LDS holds the 81x16x16 cvol tile.
// ---------------------------------------------------------------------------
extern "C" __global__ void vcn_cost_flow_kernel(
    const _Float16* __restrict__ ref16, const _Float16* __restrict__ tar16,
    const float* __restrict__ pw, float* __restrict__ out)
{
  extern __shared__ float lds_cv[];        // [NUV][16][16]

  const int lane = threadIdx.x;            // 0..31 (wave32)
  const int half = lane >> 4;
  const int m    = lane & 15;              // A-row (position) / D-col (F)

  const int blk = blockIdx.x;              // 512 blocks
  const int b   = blk >> 8;
  const int r   = blk & 255;
  const int y0  = r >> 2;
  const int x0  = (r & 3) << 4;
  const int x   = x0 + m;

  // ---- B fragment: proj_w^T (K x N), lane holds col N = m, K = 8*half + (s&7) + 16*(s>>3)
  v16h bf[4];
  #pragma unroll
  for (int q = 0; q < 4; ++q) {
    #pragma unroll
    for (int s = 0; s < 16; ++s)
      bf[q][s] = (_Float16)pw[m * C_ + q * 32 + half * 8 + (s & 7) + ((s >> 3) << 4)];
  }

  // ---- ref fragments, packed f16 (two contiguous 8-ch runs per chunk)
  const _Float16* rbase = ref16 + ((size_t)b * HW_ + y0 * W_ + x) * C_;
  v8h rlo[4], rhi[4];
  #pragma unroll
  for (int q = 0; q < 4; ++q) {
    rlo[q] = *(const v8h*)(rbase + q * 32 + half * 8);
    rhi[q] = *(const v8h*)(rbase + q * 32 + half * 8 + 16);
  }

  const v8h tenth = {(_Float16)0.1f, (_Float16)0.1f, (_Float16)0.1f,
                     (_Float16)0.1f, (_Float16)0.1f, (_Float16)0.1f,
                     (_Float16)0.1f, (_Float16)0.1f};

  // ---- 81 shifts: packed leaky(ref*tar) -> WMMA over K=128 (4 x K=32)
  for (int i = 0; i < U_; ++i) {
    for (int j = 0; j < U_; ++j) {
      const _Float16* tb =
          tar16 + (((size_t)b * PAD_ + y0 + j) * PAD_ + (x + i)) * C_;
      if (j + 1 < U_)
        __builtin_prefetch(tb + (size_t)PAD_ * C_, 0, 0);
      v8f acc = {};
      #pragma unroll
      for (int q = 0; q < 4; ++q) {
        const v8h t0 = *(const v8h*)(tb + q * 32 + half * 8);
        const v8h t1 = *(const v8h*)(tb + q * 32 + half * 8 + 16);
        v8h p0 = rlo[q] * t0;
        v8h p1 = rhi[q] * t1;
        p0 = __builtin_elementwise_max(p0, p0 * tenth);   // leaky_relu(0.1)
        p1 = __builtin_elementwise_max(p1, p1 * tenth);
        const v16h a = __builtin_shufflevector(
            p0, p1, 0, 1, 2, 3, 4, 5, 6, 7, 8, 9, 10, 11, 12, 13, 14, 15);
        acc = __builtin_amdgcn_wmma_f32_16x16x32_f16(
            false, a, false, bf[q], (short)0, acc, false, false);
      }
      const int uv = i * U_ + j;
      #pragma unroll
      for (int rr = 0; rr < 8; ++rr)
        lds_cv[(uv * 16 + (rr + 8 * half)) * 16 + m] = acc[rr];
    }
  }
  __syncthreads();

  // ---- flow_reg: each lane handles 8 of the 256 (pos, f) items
  const float inv_log49 = 1.0f / logf(49.0f);
  const float inv_log81 = 1.0f / logf(81.0f);
  for (int t = lane; t < 256; t += 32) {
    const int f  = t & 15;
    const int mm = t >> 4;

    // pass 1: argmax over 81 (first-max wins, matching jnp.argmax)
    float best = -3.4e38f; int bidx = 0;
    for (int uv = 0; uv < NUV; ++uv) {
      const float v = lds_cv[(uv * 16 + mm) * 16 + f];
      if (v > best) { best = v; bidx = uv; }
    }
    const int ib = bidx / U_, jb = bidx % U_;

    // pass 2: partition functions (window max == global max == best)
    float Zl = 0.0f, Zg = 0.0f;
    for (int i = 0; i < U_; ++i)
      for (int j = 0; j < U_; ++j) {
        const float v = lds_cv[((i * U_ + j) * 16 + mm) * 16 + f];
        const float e = __expf(v - best);
        Zg += e;
        const int di = i - ib, dj = j - jb;
        if (di >= -3 && di <= 3 && dj >= -3 && dj <= 3) Zl += e;
      }

    // pass 3: soft-argmax flow + entropies (reference clips p inside log)
    float ox = 0.0f, oy = 0.0f, Hl = 0.0f, Hg = 0.0f;
    for (int i = 0; i < U_; ++i)
      for (int j = 0; j < U_; ++j) {
        const float v = lds_cv[((i * U_ + j) * 16 + mm) * 16 + f];
        const float e = __expf(v - best);
        const float pg  = e / Zg;
        const float pgc = fminf(fmaxf(pg, 1e-9f), 1.0f - 1e-9f);
        Hg -= pg * __logf(pgc);
        const int di = i - ib, dj = j - jb;
        if (di >= -3 && di <= 3 && dj >= -3 && dj <= 3) {
          const float pl  = e / Zl;
          const float plc = fminf(fmaxf(pl, 1e-9f), 1.0f - 1e-9f);
          Hl -= pl * __logf(plc);
          ox += pl * (float)(i - MD_);
          oy += pl * (float)(j - MD_);
        }
      }

    // out1 layout: (B*F, 4, H, W) = concat([flow_h, ent_h], axis=1)
    const size_t base =
        ((size_t)(b * F_ + f) * 4) * HW_ + (size_t)y0 * W_ + (x0 + mm);
    out[base + 0 * HW_] = ox;
    out[base + 1 * HW_] = oy;
    out[base + 2 * HW_] = Hl * inv_log49;
    out[base + 3 * HW_] = Hg * inv_log81;
  }
}

// ---------------------------------------------------------------------------
// Kernel 2: bilinear backward warp of tar_feat with the f=0 flow hypothesis.
// ---------------------------------------------------------------------------
extern "C" __global__ void vcn_warp_kernel(
    const float* __restrict__ tar, const float* __restrict__ out1,
    float* __restrict__ out2)
{
  const int idx = blockIdx.x * blockDim.x + threadIdx.x;
  if (idx >= B_ * C_ * HW_) return;
  const int x = idx & 63;
  const int y = (idx >> 6) & 63;
  const int c = (idx >> 12) & 127;
  const int b = idx >> 19;

  const size_t fbase = ((size_t)(b * F_) * 4) * HW_ + (size_t)y * W_ + x;
  const float fxv = out1[fbase];
  const float fyv = out1[fbase + HW_];
  const float px = (float)x + fxv;
  const float py = (float)y + fyv;
  const bool inb = (fabsf(2.0f * px / (float)(W_ - 1) - 1.0f) < 1.0f) &&
                   (fabsf(2.0f * py / (float)(H_ - 1) - 1.0f) < 1.0f);

  const float x0f = floorf(px), y0f = floorf(py);
  const float wx = px - x0f, wy = py - y0f;
  const int x0i = (int)x0f, y0i = (int)y0f;
  const float* img = tar + (size_t)(b * C_ + c) * HW_;

  auto tap = [&](int yi, int xi, float wgt) -> float {
    const bool valid = (xi >= 0) && (xi < W_) && (yi >= 0) && (yi < H_);
    const int yc = min(max(yi, 0), H_ - 1);
    const int xc = min(max(xi, 0), W_ - 1);
    return img[yc * W_ + xc] * (valid ? wgt : 0.0f);
  };
  const float val = tap(y0i,     x0i,     (1.0f - wx) * (1.0f - wy)) +
                    tap(y0i,     x0i + 1, wx * (1.0f - wy)) +
                    tap(y0i + 1, x0i,     (1.0f - wx) * wy) +
                    tap(y0i + 1, x0i + 1, wx * wy);
  out2[idx] = inb ? val : 0.0f;
}

// ---------------------------------------------------------------------------
extern "C" void kernel_launch(void* const* d_in, const int* in_sizes, int n_in,
                              void* d_out, int out_size, void* d_ws, size_t ws_size,
                              hipStream_t stream) {
  (void)in_sizes; (void)n_in; (void)out_size; (void)ws_size;
  const float* ref = (const float*)d_in[0];
  const float* tar = (const float*)d_in[1];
  const float* pw  = (const float*)d_in[2];
  float* out1 = (float*)d_out;
  float* out2 = out1 + (size_t)B_ * F_ * 4 * HW_;

  _Float16* ref16 = (_Float16*)d_ws;                       // B*HW*C halves
  _Float16* tar16 = ref16 + (size_t)B_ * HW_ * C_;         // B*72*72*C halves

  const int nref = B_ * HW_ * C_;
  vcn_cvt_ref_kernel<<<dim3((nref + 255) / 256), dim3(256), 0, stream>>>(
      ref, ref16);
  const int ntar = B_ * PAD_ * PAD_ * C_;
  vcn_cvt_tar_kernel<<<dim3((ntar + 255) / 256), dim3(256), 0, stream>>>(
      tar, tar16);

  const size_t shmem = (size_t)NUV * 16 * 16 * sizeof(float);   // 82944 B
  vcn_cost_flow_kernel<<<dim3(B_ * H_ * (W_ / 16)), dim3(32), shmem, stream>>>(
      ref16, tar16, pw, out1);

  const int n = B_ * C_ * HW_;
  vcn_warp_kernel<<<dim3((n + 255) / 256), dim3(256), 0, stream>>>(
      tar, out1, out2);
}
